// GenNet_tanh_67791763800416
// MI455X (gfx1250) — matrix operand order repaired and checked by reading, not compiled, not run on hardware
//
#include <hip/hip_runtime.h>
#include <hip/hip_bf16.h>
#include <stdint.h>

typedef __bf16 bf16;
typedef __attribute__((ext_vector_type(16))) __bf16 v16bf;
typedef __attribute__((ext_vector_type(8)))  __bf16 v8bf;
typedef __attribute__((ext_vector_type(8)))  float  v8f;
typedef __attribute__((ext_vector_type(4))) unsigned int u32x4;
typedef __attribute__((ext_vector_type(8))) int i32x8;
typedef __attribute__((ext_vector_type(4))) int i32x4;

union ABf { v16bf v; v8bf h[2]; };

__device__ __forceinline__ float act_apply(float x, int act) {
  switch (act) {
    case 1: return x > 0.f ? x : 0.f;
    case 2: return 1.f / (1.f + __expf(-x));
    case 3: return tanhf(x);
    default: return x;
  }
}

// ---------------------------------------------------------------------------
// TDM: post one Tensor-DMA descriptor moving a rows x cols (bf16) 2-D tile
// from global (row stride = stride_elems) into LDS at lds_off, with hardware
// row padding (pad_interval_code dwords of data, then pad_amount_code+1 dwords
// of pad) so the LDS layout matches the padded tile arrays below.
// ---------------------------------------------------------------------------
__device__ __forceinline__ void tdm_load_tile(const bf16* gptr, unsigned lds_off,
                                              int rows, int cols,
                                              long long stride_elems,
                                              int pad_interval_code,
                                              int pad_amount_code) {
  unsigned long long ga = (unsigned long long)(uintptr_t)gptr;
  u32x4 g0;
  g0[0] = 1u;                                   // count=1 (valid), user mode
  g0[1] = lds_off;                              // lds_addr (bytes)
  g0[2] = (unsigned)(ga & 0xffffffffu);         // global_addr[31:0]
  g0[3] = (unsigned)((ga >> 32) & 0x1ffffffu)   // global_addr[56:32]
          | (2u << 30);                         // type = 2 ("image")
  i32x8 g1;
  g1[0] = (1 << 16)                             // data_size = 1 -> 2 bytes
        | (1 << 20)                             // pad_enable
        | (pad_interval_code << 22)             // dwords before pad
        | (pad_amount_code << 25);              // pad dwords - 1
  g1[1] = (int)(((unsigned)cols & 0xffffu) << 16);            // tensor_dim0 lo16
  g1[2] = (int)((((unsigned)cols) >> 16) |
                (((unsigned)rows & 0xffffu) << 16));          // dim0 hi | dim1 lo
  g1[3] = (int)((((unsigned)rows) >> 16) |
                (((unsigned)cols) << 16));                    // dim1 hi | tile_dim0
  g1[4] = rows;                                               // tile_dim1 | tile_dim2=0
  g1[5] = (int)(unsigned)(stride_elems & 0xffffffffll);       // dim0_stride lo32
  g1[6] = (int)(unsigned)((stride_elems >> 32) & 0xffffll);   // dim0_stride hi16
  g1[7] = 0;
  i32x4 z4;
  z4[0] = 0; z4[1] = 0; z4[2] = 0; z4[3] = 0;
  i32x8 z8;
  z8[0] = 0; z8[1] = 0; z8[2] = 0; z8[3] = 0;
  z8[4] = 0; z8[5] = 0; z8[6] = 0; z8[7] = 0;
  __builtin_amdgcn_tensor_load_to_lds(g0, g1, z4, z4, z8, 0);
}

// ---------------------------------------------------------------------------
// Generic batched GEMM:  C = act(A(bf16) @ B(bf16) + bias + Cadd)
// Block tile 128x64, 8 waves, each wave 32x32 via 2x2 WMMA 16x16x32 bf16.
// TDM-fed, double-buffered LDS: wave 0 posts descriptors for tile s+1 while
// all waves compute on tile s.
// ---------------------------------------------------------------------------
#define TM 128
#define TN 64
#define TK 32

__global__ __launch_bounds__(256)
void k_gemm_bf16(const bf16* __restrict__ A, long long strideA, int lda,
                 const bf16* __restrict__ B, long long strideB, int ldb,
                 const float* __restrict__ bias,
                 const float* __restrict__ Cadd, long long strideCadd,
                 float* __restrict__ Cf, long long strideCf,
                 bf16* __restrict__ Cb, long long strideCb,
                 int ldc, int M, int N, int K, int act) {
  (void)M; (void)N;
  // A tile rows: 32 elems (64B = 16 dwords) + 8 elems pad (16B = 4 dwords)
  // B tile rows: 64 elems (128B = 32 dwords) + 8 elems pad
  __shared__ bf16 As[2][TM][TK + 8];
  __shared__ bf16 Bs[2][TK][TN + 8];

  const int tid  = threadIdx.x;
  const int lane = tid & 31;
  const int wave = tid >> 5;
  const int wm   = wave & 3;   // 4 m-subtiles of 32 rows
  const int wn   = wave >> 2;  // 2 n-subtiles of 32 cols
  const int z    = blockIdx.z;
  const int m0   = blockIdx.y * TM;
  const int n0   = blockIdx.x * TN;

  const bf16* Abase = A + (long long)z * strideA + (long long)m0 * lda;
  const bf16* Bbase = B + (long long)z * strideB + n0;

  const int row16 = lane & 15;
  const int khalf = lane >> 4;
  const int kbase = khalf * 8;

  v8f zero8 = {0.f, 0.f, 0.f, 0.f, 0.f, 0.f, 0.f, 0.f};
  v8f acc[2][2];
  acc[0][0] = zero8; acc[0][1] = zero8; acc[1][0] = zero8; acc[1][1] = zero8;

  const unsigned lofsA[2] = {(unsigned)(uintptr_t)&As[0][0][0],
                             (unsigned)(uintptr_t)&As[1][0][0]};
  const unsigned lofsB[2] = {(unsigned)(uintptr_t)&Bs[0][0][0],
                             (unsigned)(uintptr_t)&Bs[1][0][0]};

  const int nsteps = K / TK;
  if (wave == 0) {
    tdm_load_tile(Abase, lofsA[0], TM, TK, lda, /*16 dw*/ 3, /*4 dw*/ 3);
    tdm_load_tile(Bbase, lofsB[0], TK, TN, ldb, /*32 dw*/ 4, /*4 dw*/ 3);
  }

  for (int s = 0; s < nsteps; ++s) {
    const int cur = s & 1;
    if (wave == 0) {
      if (s + 1 < nsteps) {
        const long long k1 = (long long)(s + 1) * TK;
        tdm_load_tile(Abase + k1, lofsA[cur ^ 1], TM, TK, lda, 3, 3);
        tdm_load_tile(Bbase + k1 * ldb, lofsB[cur ^ 1], TK, TN, ldb, 4, 3);
        __builtin_amdgcn_s_wait_tensorcnt(2);  // current buffer's 2 ops done
      } else {
        __builtin_amdgcn_s_wait_tensorcnt(0);
      }
    }
    __syncthreads();

    ABf afr[2], bfr[2];
#pragma unroll
    for (int i = 0; i < 2; ++i) {
      const bf16* ap = &As[cur][wm * 32 + i * 16 + row16][kbase];
      afr[i].h[0] = *(const v8bf*)(ap);
      afr[i].h[1] = *(const v8bf*)(ap + 16);
    }
#pragma unroll
    for (int j = 0; j < 2; ++j) {
      const bf16* bp = &Bs[cur][kbase][wn * 32 + j * 16 + row16];
#pragma unroll
      for (int s = 0; s < 8; ++s) {
        bfr[j].v[s] = bp[s * (TN + 8)];
        bfr[j].v[s + 8] = bp[(s + 16) * (TN + 8)];
      }
    }
#pragma unroll
    for (int i = 0; i < 2; ++i)
#pragma unroll
      for (int j = 0; j < 2; ++j)
        acc[i][j] = __builtin_amdgcn_wmma_f32_16x16x32_bf16(
            false, afr[i].v, false, bfr[j].v, (short)0, acc[i][j], false, false);
    __syncthreads();
  }

#pragma unroll
  for (int i = 0; i < 2; ++i) {
#pragma unroll
    for (int j = 0; j < 2; ++j) {
#pragma unroll
      for (int v = 0; v < 8; ++v) {
        int r = m0 + wm * 32 + i * 16 + v + 8 * khalf;
        int c = n0 + wn * 32 + j * 16 + row16;
        float x = acc[i][j][v];
        if (bias) x += bias[c];
        long long off = (long long)r * ldc + c;
        if (Cadd) x += Cadd[(long long)z * strideCadd + off];
        x = act_apply(x, act);
        if (Cf) Cf[(long long)z * strideCf + off] = x;
        if (Cb) Cb[(long long)z * strideCb + off] = (bf16)x;
      }
    }
  }
}

// ---------------------------------------------------------------------------
// Fused pairwise decoder: gram = emb@embT, scal = so@soT (K=64, WMMA),
// out[i][j] = 1 + tanh(-(sq[i]+sq[j]-2*gram)*scal)
// ---------------------------------------------------------------------------
__global__ __launch_bounds__(256)
void k_decoder(const bf16* __restrict__ Ae, const bf16* __restrict__ Bt,
               const bf16* __restrict__ Asc, const bf16* __restrict__ Bsc,
               const float* __restrict__ sq, float* __restrict__ out,
               int Nn, int Kd) {
  const int tid  = threadIdx.x;
  const int lane = tid & 31;
  const int wave = tid >> 5;
  const int wm   = wave & 3;
  const int wn   = wave >> 2;
  const int m0   = blockIdx.y * 128;
  const int n0   = blockIdx.x * 64;
  const int row16 = lane & 15;
  const int khalf = lane >> 4;
  const int kbase = khalf * 8;

  v8f zero8 = {0.f, 0.f, 0.f, 0.f, 0.f, 0.f, 0.f, 0.f};
  v8f ag[2][2], as2[2][2];
  ag[0][0] = zero8; ag[0][1] = zero8; ag[1][0] = zero8; ag[1][1] = zero8;
  as2[0][0] = zero8; as2[0][1] = zero8; as2[1][0] = zero8; as2[1][1] = zero8;

  for (int k0 = 0; k0 < Kd; k0 += 32) {
    ABf a1[2], a2[2], b1[2], b2[2];
#pragma unroll
    for (int i = 0; i < 2; ++i) {
      long long r = m0 + wm * 32 + i * 16 + row16;
      const bf16* p = Ae + r * Kd + k0 + kbase;
      a1[i].h[0] = *(const v8bf*)(p);
      a1[i].h[1] = *(const v8bf*)(p + 16);
      const bf16* q = Asc + r * Kd + k0 + kbase;
      a2[i].h[0] = *(const v8bf*)(q);
      a2[i].h[1] = *(const v8bf*)(q + 16);
    }
#pragma unroll
    for (int j = 0; j < 2; ++j) {
      long long col = n0 + wn * 32 + j * 16;
      const bf16* p = Bt + (long long)(k0 + kbase) * Nn + col + row16;
      const bf16* q = Bsc + (long long)(k0 + kbase) * Nn + col + row16;
#pragma unroll
      for (int s = 0; s < 8; ++s) {
        b1[j].v[s] = p[(long long)s * Nn];
        b1[j].v[s + 8] = p[(long long)(s + 16) * Nn];
        b2[j].v[s] = q[(long long)s * Nn];
        b2[j].v[s + 8] = q[(long long)(s + 16) * Nn];
      }
    }
#pragma unroll
    for (int i = 0; i < 2; ++i)
#pragma unroll
      for (int j = 0; j < 2; ++j) {
        ag[i][j] = __builtin_amdgcn_wmma_f32_16x16x32_bf16(
            false, a1[i].v, false, b1[j].v, (short)0, ag[i][j], false, false);
        as2[i][j] = __builtin_amdgcn_wmma_f32_16x16x32_bf16(
            false, a2[i].v, false, b2[j].v, (short)0, as2[i][j], false, false);
      }
  }

#pragma unroll
  for (int i = 0; i < 2; ++i)
#pragma unroll
    for (int j = 0; j < 2; ++j)
#pragma unroll
      for (int v = 0; v < 8; ++v) {
        int r = m0 + wm * 32 + i * 16 + v + 8 * khalf;
        int c = n0 + wn * 32 + j * 16 + row16;
        float g = ag[i][j][v];
        float s = as2[i][j][v];
        float d = -(sq[r] + sq[c] - 2.f * g);
        out[(long long)r * Nn + c] = 1.f + tanhf(d * s);
      }
}

// ---------------------------------------------------------------------------
// Elementwise / reduction helpers
// ---------------------------------------------------------------------------
__global__ void k_cvt_bf16(const float* __restrict__ src, bf16* __restrict__ dst,
                           long long n) {
  long long i = (long long)blockIdx.x * blockDim.x + threadIdx.x;
  if (i < n) dst[i] = (bf16)src[i];
}

__global__ void k_zero_u32(uint32_t* __restrict__ p, long long n) {
  long long i = (long long)blockIdx.x * blockDim.x + threadIdx.x;
  if (i < n) p[i] = 0u;
}

__global__ void k_copy2d_bf16(const bf16* __restrict__ src, int sld, long long sstride,
                              bf16* __restrict__ dst, int dld, long long dstride,
                              int rows, int cols) {
  int idx = blockIdx.x * blockDim.x + threadIdx.x;
  int z = blockIdx.z;
  if (idx >= rows * cols) return;
  int r = idx / cols, c = idx - r * cols;
  dst[(long long)z * dstride + (long long)r * dld + c] =
      src[(long long)z * sstride + (long long)r * sld + c];
}

__global__ void k_transpose_bf16(const bf16* __restrict__ src, long long sstride,
                                 bf16* __restrict__ dst, long long dstride,
                                 int R, int C) {
  int idx = blockIdx.x * blockDim.x + threadIdx.x;
  int z = blockIdx.z;
  if (idx >= R * C) return;
  int r = idx / C, c = idx - r * C;
  dst[(long long)z * dstride + (long long)c * R + r] =
      src[(long long)z * sstride + (long long)r * C + c];
}

__global__ __launch_bounds__(256)
void k_l2norm_rows(const float* __restrict__ in, bf16* __restrict__ out, int C) {
  __shared__ float red[256];
  int r = blockIdx.x;
  float x = in[(long long)r * C + threadIdx.x];  // C == blockDim.x == 256
  red[threadIdx.x] = x * x;
  __syncthreads();
  for (int s = 128; s > 0; s >>= 1) {
    if ((int)threadIdx.x < s) red[threadIdx.x] += red[threadIdx.x + s];
    __syncthreads();
  }
  float inv = 1.f / fmaxf(sqrtf(red[0]), 1e-12f);
  out[(long long)r * C + threadIdx.x] = (bf16)(x * inv);
}

__global__ __launch_bounds__(256)
void k_softmax_mix(const float* __restrict__ att, const float* __restrict__ al,
                   const float* __restrict__ lambd_p, bf16* __restrict__ out, int Ncol) {
  __shared__ float red[256];
  int r = blockIdx.x;
  const float* arow = att + (long long)r * Ncol;
  const float* lrow = al + (long long)r * Ncol;
  float lam = lambd_p[0];
  float mx = -3.4e38f;
  for (int c = threadIdx.x; c < Ncol; c += 256) mx = fmaxf(mx, arow[c]);
  red[threadIdx.x] = mx;
  __syncthreads();
  for (int s = 128; s > 0; s >>= 1) {
    if ((int)threadIdx.x < s) red[threadIdx.x] = fmaxf(red[threadIdx.x], red[threadIdx.x + s]);
    __syncthreads();
  }
  mx = red[0];
  __syncthreads();
  float sum = 0.f;
  for (int c = threadIdx.x; c < Ncol; c += 256) sum += __expf(arow[c] - mx);
  red[threadIdx.x] = sum;
  __syncthreads();
  for (int s = 128; s > 0; s >>= 1) {
    if ((int)threadIdx.x < s) red[threadIdx.x] += red[threadIdx.x + s];
    __syncthreads();
  }
  float inv = 1.f / red[0];
  bf16* orow = out + (long long)r * Ncol;
  for (int c = threadIdx.x; c < Ncol; c += 256)
    orow[c] = (bf16)(lam * lrow[c] + (1.f - lam) * __expf(arow[c] - mx) * inv);
}

// r-gate * h  -> bf16
__global__ void k_rh(const float* __restrict__ gates, const float* __restrict__ h,
                     bf16* __restrict__ out, int total) {
  int i = blockIdx.x * blockDim.x + threadIdx.x;
  if (i >= total) return;
  int r = i >> 8, c = i & 255;
  out[i] = (bf16)(gates[(long long)r * 512 + c] * h[i]);
}

// rnn = (1-u)*h + u*c
__global__ void k_rnn(const float* __restrict__ gates, const float* __restrict__ h,
                      const float* __restrict__ cc, bf16* __restrict__ out, int total) {
  int i = blockIdx.x * blockDim.x + threadIdx.x;
  if (i >= total) return;
  int r = i >> 8, c = i & 255;
  float u = gates[(long long)r * 512 + 256 + c];
  out[i] = (bf16)((1.f - u) * h[i] + u * cc[i]);
}

__global__ __launch_bounds__(256)
void k_colnorm(const float* __restrict__ emb, float* __restrict__ cn, int R, int C) {
  __shared__ float red[256];
  int c = blockIdx.x;
  float s = 0.f;
  for (int r = threadIdx.x; r < R; r += 256) {
    float v = emb[(long long)r * C + c];
    s += v * v;
  }
  red[threadIdx.x] = s;
  __syncthreads();
  for (int st = 128; st > 0; st >>= 1) {
    if ((int)threadIdx.x < st) red[threadIdx.x] += red[threadIdx.x + st];
    __syncthreads();
  }
  if (threadIdx.x == 0) cn[c] = 1.f / fmaxf(sqrtf(red[0]), 1e-12f);
}

__global__ void k_scale_cols(float* __restrict__ emb, const float* __restrict__ cn,
                             bf16* __restrict__ out_bf, int total, int C) {
  int i = blockIdx.x * blockDim.x + threadIdx.x;
  if (i >= total) return;
  int c = i % C;
  float v = emb[i] * cn[c];
  emb[i] = v;
  out_bf[i] = (bf16)v;
}

__global__ void k_rowsumsq(const float* __restrict__ emb, float* __restrict__ sq,
                           int R, int C) {
  int r = blockIdx.x * blockDim.x + threadIdx.x;
  if (r >= R) return;
  float s = 0.f;
  for (int c = 0; c < C; ++c) {
    float v = emb[(long long)r * C + c];
    s += v * v;
  }
  sq[r] = s;
}

// ---------------------------------------------------------------------------
extern "C" void kernel_launch(void* const* d_in, const int* in_sizes, int n_in,
                              void* d_out, int out_size, void* d_ws, size_t ws_size,
                              hipStream_t stream) {
  (void)in_sizes; (void)n_in; (void)out_size; (void)ws_size;
  const int N = 2048, T = 4, FIN = 128, HF = 256, H = 256, DZ = 128;
  const int GIN = 384, EIN = 512, E1 = 256, E2 = 64;

  const float* sup     = (const float*)d_in[0];
  const float* feat    = (const float*)d_in[1];
  const float* noise   = (const float*)d_in[2];
  const float* align_m = (const float*)d_in[3];
  const float* lambd   = (const float*)d_in[4];
  const float* W_fem   = (const float*)d_in[5];  const float* b_fem   = (const float*)d_in[6];
  const float* W_gnn   = (const float*)d_in[7];  const float* b_gnn   = (const float*)d_in[8];
  const float* Wg_x    = (const float*)d_in[9];  const float* Wg_h    = (const float*)d_in[10];
  const float* bg      = (const float*)d_in[11];
  const float* Wc_x    = (const float*)d_in[12]; const float* Wc_h    = (const float*)d_in[13];
  const float* bc      = (const float*)d_in[14];
  const float* W_att   = (const float*)d_in[15]; const float* b_att   = (const float*)d_in[16];
  const float* W_emb1  = (const float*)d_in[17]; const float* b_emb1  = (const float*)d_in[18];
  const float* W_emb2  = (const float*)d_in[19]; const float* b_emb2  = (const float*)d_in[20];
  const float* W_scal1 = (const float*)d_in[21]; const float* b_scal1 = (const float*)d_in[22];
  const float* W_scal2 = (const float*)d_in[23]; const float* b_scal2 = (const float*)d_in[24];

  char* ws = (char*)d_ws;
  size_t off = 0;
  auto alloc = [&](size_t bytes) -> char* {
    char* p = ws + off;
    off = (off + bytes + 255) & ~(size_t)255;
    return p;
  };

  bf16* sup_bf   = (bf16*)alloc((size_t)T * N * N * 2);
  bf16* feat_bf  = (bf16*)alloc((size_t)(T + 1) * N * FIN * 2);
  bf16* noise_bf = (bf16*)alloc((size_t)T * N * DZ * 2);
  bf16* Wfem_bf  = (bf16*)alloc((size_t)FIN * HF * 2);
  bf16* Wgnn_bf  = (bf16*)alloc((size_t)GIN * H * 2);
  bf16* Wgx_bf   = (bf16*)alloc((size_t)H * 2 * H * 2);
  bf16* Wgh_bf   = (bf16*)alloc((size_t)H * 2 * H * 2);
  bf16* Wcx_bf   = (bf16*)alloc((size_t)H * H * 2);
  bf16* Wch_bf   = (bf16*)alloc((size_t)H * H * 2);
  bf16* Watt_bf  = (bf16*)alloc((size_t)HF * H * 2);
  bf16* We1_bf   = (bf16*)alloc((size_t)EIN * E1 * 2);
  bf16* We2_bf   = (bf16*)alloc((size_t)E1 * E2 * 2);
  bf16* Ws1_bf   = (bf16*)alloc((size_t)EIN * E1 * 2);
  bf16* Ws2_bf   = (bf16*)alloc((size_t)E1 * E2 * 2);
  bf16* fem_bf   = (bf16*)alloc((size_t)(T + 1) * N * HF * 2);
  bf16* edm_bf   = (bf16*)alloc((size_t)T * N * GIN * 2);
  bf16* agg_bf   = (bf16*)alloc((size_t)T * N * GIN * 2);
  float* gnn_f32 = (float*)alloc((size_t)T * N * H * 4);
  bf16* g_bf     = (bf16*)alloc((size_t)T * N * H * 2);
  bf16* fm_bf    = (bf16*)alloc((size_t)(T + 1) * N * H * 2);
  bf16* fmT_bf   = (bf16*)alloc((size_t)(T + 1) * H * N * 2);
  float* h_f32   = (float*)alloc((size_t)N * H * 4);
  bf16* h_bf     = (bf16*)alloc((size_t)N * H * 2);
  float* tmp1    = (float*)alloc((size_t)N * 2 * H * 4);
  float* gates   = (float*)alloc((size_t)N * 2 * H * 4);
  bf16* rh_bf    = (bf16*)alloc((size_t)N * H * 2);
  float* tmp2    = (float*)alloc((size_t)N * H * 4);
  float* cc_f32  = (float*)alloc((size_t)N * H * 4);
  bf16* rnn_bf   = (bf16*)alloc((size_t)N * H * 2);
  float* att_f32 = (float*)alloc((size_t)N * N * 4);
  bf16* supal_bf = (bf16*)alloc((size_t)N * N * 2);
  bf16* ecat_bf  = (bf16*)alloc((size_t)N * EIN * 2);
  bf16* e1_bf    = (bf16*)alloc((size_t)N * E1 * 2);
  float* emb_f32 = (float*)alloc((size_t)N * E2 * 4);
  bf16* emb_bf   = (bf16*)alloc((size_t)N * E2 * 2);
  bf16* embT_bf  = (bf16*)alloc((size_t)E2 * N * 2);
  bf16* s1_bf    = (bf16*)alloc((size_t)N * E1 * 2);
  bf16* so_bf    = (bf16*)alloc((size_t)N * E2 * 2);
  bf16* soT_bf   = (bf16*)alloc((size_t)E2 * N * 2);
  float* cn_f32  = (float*)alloc((size_t)E2 * 4);
  float* sq_f32  = (float*)alloc((size_t)N * 4);

  dim3 blk(256);
  auto cvt = [&](const float* s, bf16* d, long long n) {
    k_cvt_bf16<<<dim3((unsigned)((n + 255) / 256)), blk, 0, stream>>>(s, d, n);
  };
  auto gemm = [&](const bf16* A, long long sA, int lda, const bf16* B, long long sB,
                  int ldb, const float* bias, const float* Cadd, long long sCadd,
                  float* Cf, long long sCf, bf16* Cb, long long sCb, int ldc,
                  int M, int Ncol, int K, int act, int batch) {
    dim3 g((unsigned)(Ncol / 64), (unsigned)(M / 128), (unsigned)batch);
    k_gemm_bf16<<<g, blk, 0, stream>>>(A, sA, lda, B, sB, ldb, bias, Cadd, sCadd,
                                       Cf, sCf, Cb, sCb, ldc, M, Ncol, K, act);
  };

  // ---- precision conversions ----
  cvt(sup, sup_bf, (long long)T * N * N);
  cvt(feat, feat_bf, (long long)(T + 1) * N * FIN);
  cvt(noise, noise_bf, (long long)T * N * DZ);
  cvt(W_fem, Wfem_bf, FIN * HF);      cvt(W_gnn, Wgnn_bf, GIN * H);
  cvt(Wg_x, Wgx_bf, H * 2 * H);       cvt(Wg_h, Wgh_bf, H * 2 * H);
  cvt(Wc_x, Wcx_bf, H * H);           cvt(Wc_h, Wch_bf, H * H);
  cvt(W_att, Watt_bf, HF * H);
  cvt(W_emb1, We1_bf, EIN * E1);      cvt(W_emb2, We2_bf, E1 * E2);
  cvt(W_scal1, Ws1_bf, EIN * E1);     cvt(W_scal2, Ws2_bf, E1 * E2);

  // ---- FEM: fem = relu(feat @ W_fem + b) (batch T+1) ----
  gemm(feat_bf, (long long)N * FIN, FIN, Wfem_bf, 0, HF, b_fem, nullptr, 0,
       nullptr, 0, fem_bf, (long long)N * HF, HF, N, HF, FIN, 1, T + 1);

  // ---- edm_in = concat(fem[:T], noise) ----
  k_copy2d_bf16<<<dim3((unsigned)((N * HF + 255) / 256), 1, T), blk, 0, stream>>>(
      fem_bf, HF, (long long)N * HF, edm_bf, GIN, (long long)N * GIN, N, HF);
  k_copy2d_bf16<<<dim3((unsigned)((N * DZ + 255) / 256), 1, T), blk, 0, stream>>>(
      noise_bf, DZ, (long long)N * DZ, edm_bf + HF, GIN, (long long)N * GIN, N, DZ);

  // ---- agg = sup @ edm_in (batch T) ----
  gemm(sup_bf, (long long)N * N, N, edm_bf, (long long)N * GIN, GIN, nullptr,
       nullptr, 0, nullptr, 0, agg_bf, (long long)N * GIN, GIN, N, GIN, N, 0, T);

  // ---- g = l2norm(relu(agg @ W_gnn + b)) ----
  gemm(agg_bf, (long long)N * GIN, GIN, Wgnn_bf, 0, H, b_gnn, nullptr, 0,
       gnn_f32, (long long)N * H, nullptr, 0, H, N, H, GIN, 1, T);
  k_l2norm_rows<<<dim3((unsigned)(T * N)), blk, 0, stream>>>(gnn_f32, g_bf, H);

  // ---- fm = fem @ W_att + b (batch T+1), + transposes for attention B ----
  gemm(fem_bf, (long long)N * HF, HF, Watt_bf, 0, H, b_att, nullptr, 0,
       nullptr, 0, fm_bf, (long long)N * H, H, N, H, HF, 0, T + 1);
  k_transpose_bf16<<<dim3((unsigned)((N * H + 255) / 256), 1, T + 1), blk, 0, stream>>>(
      fm_bf, (long long)N * H, fmT_bf, (long long)N * H, N, H);

  // ---- h0 = 0 ----
  k_zero_u32<<<dim3((unsigned)((N * H + 255) / 256)), blk, 0, stream>>>(
      (uint32_t*)h_f32, (long long)N * H);
  k_zero_u32<<<dim3((unsigned)((N * H / 2 + 255) / 256)), blk, 0, stream>>>(
      (uint32_t*)h_bf, (long long)N * H / 2);

  // ---- scan over T steps ----
  for (int t = 0; t < T; ++t) {
    const bf16* g_t = g_bf + (long long)t * N * H;
    // gates = sigmoid(g@Wg_x + h@Wg_h + bg)
    gemm(g_t, 0, H, Wgx_bf, 0, 2 * H, bg, nullptr, 0, tmp1, 0, nullptr, 0,
         2 * H, N, 2 * H, H, 0, 1);
    gemm(h_bf, 0, H, Wgh_bf, 0, 2 * H, nullptr, tmp1, 0, gates, 0, nullptr, 0,
         2 * H, N, 2 * H, H, 2, 1);
    // rh = r * h
    k_rh<<<dim3((unsigned)(N * H / 256)), blk, 0, stream>>>(gates, h_f32, rh_bf, N * H);
    // c = tanh(g@Wc_x + rh@Wc_h + bc)
    gemm(g_t, 0, H, Wcx_bf, 0, H, bc, nullptr, 0, tmp2, 0, nullptr, 0,
         H, N, H, H, 0, 1);
    gemm(rh_bf, 0, H, Wch_bf, 0, H, nullptr, tmp2, 0, cc_f32, 0, nullptr, 0,
         H, N, H, H, 3, 1);
    // rnn = (1-u)*h + u*c
    k_rnn<<<dim3((unsigned)(N * H / 256)), blk, 0, stream>>>(gates, h_f32, cc_f32,
                                                             rnn_bf, N * H);
    // att = fdst @ fsrc^T ; sup_al = lam*align + (1-lam)*softmax(att)
    gemm(fm_bf + (long long)(t + 1) * N * H, 0, H, fmT_bf + (long long)t * H * N, 0, N,
         nullptr, nullptr, 0, att_f32, 0, nullptr, 0, N, N, N, H, 0, 1);
    k_softmax_mix<<<dim3((unsigned)N), blk, 0, stream>>>(
        att_f32, align_m + (long long)t * N * N, lambd, supal_bf, N);
    // h = sup_al @ rnn
    gemm(supal_bf, 0, N, rnn_bf, 0, H, nullptr, nullptr, 0, h_f32, 0, h_bf, 0,
         H, N, H, N, 0, 1);
  }

  // ---- EAM ----
  k_copy2d_bf16<<<dim3((unsigned)((N * H + 255) / 256), 1, 1), blk, 0, stream>>>(
      h_bf, H, 0, ecat_bf, EIN, 0, N, H);
  k_copy2d_bf16<<<dim3((unsigned)((N * HF + 255) / 256), 1, 1), blk, 0, stream>>>(
      fem_bf + (long long)T * N * HF, HF, 0, ecat_bf + HF, EIN, 0, N, HF);

  // emb head
  gemm(ecat_bf, 0, EIN, We1_bf, 0, E1, b_emb1, nullptr, 0, nullptr, 0, e1_bf, 0,
       E1, N, E1, EIN, 3, 1);
  gemm(e1_bf, 0, E1, We2_bf, 0, E2, b_emb2, nullptr, 0, emb_f32, 0, nullptr, 0,
       E2, N, E2, E1, 3, 1);
  k_colnorm<<<dim3((unsigned)E2), blk, 0, stream>>>(emb_f32, cn_f32, N, E2);
  k_scale_cols<<<dim3((unsigned)(N * E2 / 256)), blk, 0, stream>>>(
      emb_f32, cn_f32, emb_bf, N * E2, E2);
  k_transpose_bf16<<<dim3((unsigned)((N * E2 + 255) / 256), 1, 1), blk, 0, stream>>>(
      emb_bf, 0, embT_bf, 0, N, E2);
  k_rowsumsq<<<dim3((unsigned)((N + 255) / 256)), blk, 0, stream>>>(emb_f32, sq_f32, N, E2);

  // scaling head
  gemm(ecat_bf, 0, EIN, Ws1_bf, 0, E1, b_scal1, nullptr, 0, nullptr, 0, s1_bf, 0,
       E1, N, E1, EIN, 2, 1);
  gemm(s1_bf, 0, E1, Ws2_bf, 0, E2, b_scal2, nullptr, 0, nullptr, 0, so_bf, 0,
       E2, N, E2, E1, 2, 1);
  k_transpose_bf16<<<dim3((unsigned)((N * E2 + 255) / 256), 1, 1), blk, 0, stream>>>(
      so_bf, 0, soT_bf, 0, N, E2);

  // fused decoder: out = 1 + tanh(-(sq_i + sq_j - 2*emb@embT) * (so@soT))
  k_decoder<<<dim3((unsigned)(N / 64), (unsigned)(N / 128), 1), blk, 0, stream>>>(
      emb_bf, embT_bf, so_bf, soT_bf, sq_f32, (float*)d_out, N, E2);
}
